// TraditionAttention_28389733826868
// MI455X (gfx1250) — matrix operand order repaired and checked by reading, not compiled, not run on hardware
//
#include <hip/hip_runtime.h>
#include <hip/hip_bf16.h>

typedef __bf16 bf16;
typedef __attribute__((ext_vector_type(16))) __bf16 v16bf;
typedef __attribute__((ext_vector_type(8)))  float  v8f;

union FragAB { v16bf v; uint4 q[2]; };

__device__ __forceinline__ v8f wmma_bf16(const FragAB& a, const FragAB& b, v8f c) {
  return __builtin_amdgcn_wmma_f32_16x16x32_bf16(false, a.v, false, b.v,
                                                 (short)0, c, false, false);
}

// Async global -> LDS 128-bit copy (ASYNCcnt-tracked), GV addressing mode.
__device__ __forceinline__ void async_b128(unsigned int lds_off, const void* gaddr) {
  asm volatile("global_load_async_to_lds_b128 %0, %1, off"
               :: "v"(lds_off), "v"(gaddr) : "memory");
}

__device__ __forceinline__ void wait_async0() {
#if __has_builtin(__builtin_amdgcn_s_wait_asynccnt)
  __builtin_amdgcn_s_wait_asynccnt(0);
#else
  asm volatile("s_wait_asynccnt 0x0" ::: "memory");
#endif
}

// ---------------------------------------------------------------------------
// Stage 1: NCHW fp32 -> [B*HW, C] bf16 (masked + unmasked), 32x32 LDS tiles
// ---------------------------------------------------------------------------
__global__ __launch_bounds__(256) void prep_transpose(
    const float* __restrict__ x, const float* __restrict__ mask,
    bf16* __restrict__ Xt, bf16* __restrict__ Xmt)
{
  __shared__ float tile[32][33];
  const int c0 = blockIdx.x * 32, p0 = blockIdx.y * 32, b = blockIdx.z;
  const int t = threadIdx.x;
  {
    const int i = t >> 3, j4 = (t & 7) * 4;
    const float4 v = *(const float4*)(x + ((long)(b * 512 + c0 + i)) * 4096 + p0 + j4);
    tile[i][j4 + 0] = v.x; tile[i][j4 + 1] = v.y;
    tile[i][j4 + 2] = v.z; tile[i][j4 + 3] = v.w;
  }
  __syncthreads();
  const int jj = t >> 3, i4 = (t & 7) * 4;
  const float mval = mask[(long)b * 4096 + p0 + jj];
  const long orow = ((long)b * 4096 + p0 + jj) * 512 + c0 + i4;
#pragma unroll
  for (int u = 0; u < 4; ++u) {
    const float v = tile[i4 + u][jj];
    Xt[orow + u]  = (bf16)v;
    Xmt[orow + u] = (bf16)(v * mval);
  }
}

// ---------------------------------------------------------------------------
// Stage 2: fp32 weights -> bf16 (row-major kept: C = A * W^T reads W rows)
// ---------------------------------------------------------------------------
__global__ __launch_bounds__(256) void pack_weights(
    const float* __restrict__ a, const float* __restrict__ b,
    const float* __restrict__ c, const float* __restrict__ d,
    bf16* __restrict__ out)
{
  const int idx = blockIdx.x * 256 + threadIdx.x;      // 4 * 262144 total
  const float* srcs[4] = {a, b, c, d};
  out[idx] = (bf16)srcs[idx >> 18][idx & 262143];
}

// ---------------------------------------------------------------------------
// Stage 3/6/7: WMMA GEMM  C = A[M,K] * W[N,K]^T
// 256 threads = 8 waves (4x2), macro tile 128x64, BK = 64 (2 WMMA K-steps).
// Double-buffered LDS filled by async global->LDS; one s_wait_asynccnt +
// one barrier per staged tile; 8 WMMAs per barrier per wave-pair tile.
// MODE 0: bf16 store to Cb.  MODE 1: fp32 scatter to NCHW (B,512,64,64).
// Batched via z: off = (z>>3)*Out + (z&7)*In per operand.
// ---------------------------------------------------------------------------
template <int MODE>
__global__ __launch_bounds__(256) void gemm_xwT(
    const bf16* __restrict__ A, const bf16* __restrict__ W,
    bf16* __restrict__ Cb, float* __restrict__ Cf,
    int M, int N, int K, int lda, int ldw, int ldc,
    long aOut, long aIn, long wOut, long wIn, long cOut, long cIn)
{
  constexpr int AS = 80, BS = 80;                      // padded LDS row strides
  __shared__ bf16 As[2][128 * AS];
  __shared__ bf16 Bs[2][64 * BS];

  const int z = blockIdx.z;
  A += (long)(z >> 3) * aOut + (long)(z & 7) * aIn;
  W += (long)(z >> 3) * wOut + (long)(z & 7) * wIn;
  const long cOff = (long)(z >> 3) * cOut + (long)(z & 7) * cIn;

  const int m0 = blockIdx.x * 128, n0 = blockIdx.y * 64;
  const int t = threadIdx.x, wave = t >> 5, lane = t & 31;
  const int wm = (wave >> 1) * 32, wn = (wave & 1) * 32;
  const int lrow = lane & 15, lkg = lane >> 4;

  // async fill mapping: A tile 128x64 -> 4 b128/thread; B tile 64x64 -> 2/thread
  const int arow = t >> 1, ahalf = t & 1;
  const int brow = t >> 2, bq = t & 3;
  const unsigned int asBase = (unsigned int)(unsigned long long)&As[0][0];
  const unsigned int bsBase = (unsigned int)(unsigned long long)&Bs[0][0];

  auto issueTile = [&](int k0, int buf) {
    const bf16* ga = A + (long)(m0 + arow) * lda + k0 + ahalf * 32;
    const unsigned int la =
        asBase + (unsigned int)buf * (128 * AS * 2) + arow * (AS * 2) + ahalf * 64;
#pragma unroll
    for (int u = 0; u < 4; ++u) async_b128(la + u * 16, ga + u * 8);
    const bf16* gb = W + (long)(n0 + brow) * ldw + k0 + bq * 16;
    const unsigned int lb =
        bsBase + (unsigned int)buf * (64 * BS * 2) + brow * (BS * 2) + bq * 32;
#pragma unroll
    for (int u = 0; u < 2; ++u) async_b128(lb + u * 16, gb + u * 8);
  };

  v8f acc[2][2] = {};

  issueTile(0, 0);
  const int KT = K >> 6;
  for (int kt = 0; kt < KT; ++kt) {
    wait_async0();
    __syncthreads();                     // tile kt resident; prev reads done
    if (kt + 1 < KT) issueTile((kt + 1) << 6, (kt + 1) & 1);

    const bf16* as = &As[kt & 1][0];
    const bf16* bs = &Bs[kt & 1][0];
#pragma unroll
    for (int ks = 0; ks < 2; ++ks) {
      FragAB af[2], bf_[2];
#pragma unroll
      for (int mt = 0; mt < 2; ++mt) {
        const bf16* p = &as[(wm + mt * 16 + lrow) * AS + ks * 32 + lkg * 8];
        af[mt].q[0] = *(const uint4*)p;
        af[mt].q[1] = *(const uint4*)(p + 16);
      }
#pragma unroll
      for (int nt = 0; nt < 2; ++nt) {
        const bf16* p = &bs[(wn + nt * 16 + lrow) * BS + ks * 32 + lkg * 8];
        bf_[nt].q[0] = *(const uint4*)p;
        bf_[nt].q[1] = *(const uint4*)(p + 16);
      }
#pragma unroll
      for (int mt = 0; mt < 2; ++mt)
#pragma unroll
        for (int nt = 0; nt < 2; ++nt)
          acc[mt][nt] = wmma_bf16(af[mt], bf_[nt], acc[mt][nt]);
    }
  }

  const int en = lane & 15, emb = (lane >> 4) * 8;
#pragma unroll
  for (int mt = 0; mt < 2; ++mt)
#pragma unroll
    for (int nt = 0; nt < 2; ++nt) {
      const int gn = n0 + wn + nt * 16 + en;
#pragma unroll
      for (int v = 0; v < 8; ++v) {
        const int gm = m0 + wm + mt * 16 + emb + v;
        const float val = acc[mt][nt][v];
        if (MODE == 0) {
          Cb[cOff + (long)gm * ldc + gn] = (bf16)val;
        } else {
          // gm = b*4096 + pix ; out[b, gn, pix] in NCHW fp32
          Cf[((long)((gm >> 12) * 512 + gn)) * 4096 + (gm & 4095)] = val;
        }
      }
    }
}

// ---------------------------------------------------------------------------
// Stage 4: column L2 norms over HW=4096: out[b*512+col] = sqrt(sum x^2)
// ---------------------------------------------------------------------------
__global__ __launch_bounds__(256) void col_l2norm(
    const bf16* __restrict__ X, float* __restrict__ out)
{
  const int col = blockIdx.x & 511, b = blockIdx.x >> 9;
  const bf16* p = X + (long)b * 4096 * 512 + col;
  float s = 0.f;
  for (int n = threadIdx.x; n < 4096; n += 256) {
    const float v = (float)p[(long)n * 512];
    s += v * v;
  }
  __shared__ float red[256];
  red[threadIdx.x] = s;
  __syncthreads();
  for (int off = 128; off; off >>= 1) {
    if (threadIdx.x < off) red[threadIdx.x] += red[threadIdx.x + off];
    __syncthreads();
  }
  if (threadIdx.x == 0) out[blockIdx.x] = sqrtf(red[0]);
}

// ---------------------------------------------------------------------------
// Stage 5: per (b,h): S = Q_h^T K_h (64x64, K-dim 4096) via WMMA, then fold
// scale / L2 norms and softmax; emit P bf16 [z][64][64].
// Transposed LDS tiles built with packed b32 stores (2 n-values per store).
// ---------------------------------------------------------------------------
__global__ __launch_bounds__(128) void attn_softmax(
    const bf16* __restrict__ Q, const bf16* __restrict__ Kt,
    const float* __restrict__ nq, const float* __restrict__ nk,
    bf16* __restrict__ P)
{
  constexpr int ST = 48;
  __shared__ bf16 Qs[64 * ST];     // Qs[c][n]  (transposed tile)
  __shared__ bf16 Ks[64 * ST];     // Ks[d][n]
  __shared__ float Sf[64 * 64];

  const int z = blockIdx.x, b = z >> 3, h = z & 7;
  const bf16* Qh = Q  + (long)b * 4096 * 512 + h * 64;
  const bf16* Kh = Kt + (long)b * 4096 * 512 + h * 64;

  const int t = threadIdx.x, wave = t >> 5, lane = t & 31;
  const int lrow = lane & 15, lkg = lane >> 4;
  const int pr = t & 15;           // n-pair: n = 2*pr, 2*pr+1
  const int g  = t >> 4;           // channel group: cols g*8 .. g*8+7

  v8f acc[4] = {};

  for (int n0 = 0; n0 < 4096; n0 += 32) {
    const uint4 qa = *(const uint4*)(Qh + (long)(n0 + 2 * pr)     * 512 + g * 8);
    const uint4 qb = *(const uint4*)(Qh + (long)(n0 + 2 * pr + 1) * 512 + g * 8);
    const uint4 ka = *(const uint4*)(Kh + (long)(n0 + 2 * pr)     * 512 + g * 8);
    const uint4 kb = *(const uint4*)(Kh + (long)(n0 + 2 * pr + 1) * 512 + g * 8);
    __syncthreads();
    const bf16* qal = (const bf16*)&qa; const bf16* qbl = (const bf16*)&qb;
    const bf16* kal = (const bf16*)&ka; const bf16* kbl = (const bf16*)&kb;
#pragma unroll
    for (int j = 0; j < 8; ++j) {
      union { bf16 h2[2]; unsigned int u; } pq, pk2;
      pq.h2[0]  = qal[j]; pq.h2[1]  = qbl[j];
      pk2.h2[0] = kal[j]; pk2.h2[1] = kbl[j];
      *(unsigned int*)&Qs[(g * 8 + j) * ST + 2 * pr] = pq.u;
      *(unsigned int*)&Ks[(g * 8 + j) * ST + 2 * pr] = pk2.u;
    }
    __syncthreads();

    FragAB a;
    const bf16* pa = &Qs[(wave * 16 + lrow) * ST + lkg * 8];
    a.q[0] = *(const uint4*)pa;
    a.q[1] = *(const uint4*)(pa + 16);
#pragma unroll
    for (int nt = 0; nt < 4; ++nt) {
      FragAB bb;
      const bf16* pb = &Ks[(nt * 16 + lrow) * ST + lkg * 8];
      bb.q[0] = *(const uint4*)pb;
      bb.q[1] = *(const uint4*)(pb + 16);
      acc[nt] = wmma_bf16(a, bb, acc[nt]);
    }
  }

  const int en = lane & 15, emb = (lane >> 4) * 8;
#pragma unroll
  for (int nt = 0; nt < 4; ++nt)
#pragma unroll
    for (int v = 0; v < 8; ++v)
      Sf[(wave * 16 + emb + v) * 64 + nt * 16 + en] = acc[nt][v];
  __syncthreads();

  if (t < 64) {
    const int r = t;
    const float qn = fmaxf(nq[b * 512 + h * 64 + r], 1e-12f);
    float mx = -1e30f;
    for (int c = 0; c < 64; ++c) {
      const float kn = fmaxf(nk[b * 512 + h * 64 + c], 1e-12f);
      const float s = Sf[r * 64 + c] * 0.125f / (qn * kn);
      Sf[r * 64 + c] = s;
      mx = fmaxf(mx, s);
    }
    float sum = 0.f;
    for (int c = 0; c < 64; ++c) {
      const float e = __expf(Sf[r * 64 + c] - mx);
      Sf[r * 64 + c] = e;
      sum += e;
    }
    const float inv = 1.f / sum;
    for (int c = 0; c < 64; ++c)
      P[((long)z * 64 + r) * 64 + c] = (bf16)(Sf[r * 64 + c] * inv);
  }
}

// ---------------------------------------------------------------------------
extern "C" void kernel_launch(void* const* d_in, const int* in_sizes, int n_in,
                              void* d_out, int out_size, void* d_ws, size_t ws_size,
                              hipStream_t stream)
{
  const float* x    = (const float*)d_in[0];
  const float* mask = (const float*)d_in[1];
  const float* Wq   = (const float*)d_in[2];
  const float* Wk   = (const float*)d_in[3];
  const float* Wv   = (const float*)d_in[4];
  const float* Wp   = (const float*)d_in[5];
  float* out = (float*)d_out;

  char* ws = (char*)d_ws;
  size_t off = 0;
  auto alloc = [&](size_t bytes) -> void* {
    void* p = ws + off;
    off += (bytes + 255) & ~(size_t)255;
    return p;
  };

  const size_t BIG = (size_t)32768 * 512 * sizeof(bf16);   // 32 MB
  bf16* Xt   = (bf16*)alloc(BIG);
  bf16* Xmt  = (bf16*)alloc(BIG);
  bf16* Qb   = (bf16*)alloc(BIG);
  bf16* Kb   = (bf16*)alloc(BIG);
  bf16* Vb   = (bf16*)alloc(BIG);
  bf16* Wb   = (bf16*)alloc((size_t)4 * 512 * 512 * sizeof(bf16));
  float* nq  = (float*)alloc((size_t)8 * 512 * sizeof(float));
  float* nk  = (float*)alloc((size_t)8 * 512 * sizeof(float));
  bf16* P    = (bf16*)alloc((size_t)64 * 64 * 64 * sizeof(bf16));
  bf16* Y    = Xmt;                                        // Xmt dead after Q,K GEMMs
  if (off > ws_size) return;

  // 1) transpose + mask + pack
  prep_transpose<<<dim3(16, 128, 8), 256, 0, stream>>>(x, mask, Xt, Xmt);
  // 2) weights -> bf16
  pack_weights<<<4096, 256, 0, stream>>>(Wq, Wk, Wv, Wp, Wb);
  // 3) Q, K, V projections: [32768,512] x [512,512]^T
  gemm_xwT<0><<<dim3(256, 8, 1), 256, 0, stream>>>(
      Xmt, Wb + 0 * 262144, Qb, nullptr, 32768, 512, 512, 512, 512, 512,
      0, 0, 0, 0, 0, 0);
  gemm_xwT<0><<<dim3(256, 8, 1), 256, 0, stream>>>(
      Xmt, Wb + 1 * 262144, Kb, nullptr, 32768, 512, 512, 512, 512, 512,
      0, 0, 0, 0, 0, 0);
  gemm_xwT<0><<<dim3(256, 8, 1), 256, 0, stream>>>(
      Xt,  Wb + 2 * 262144, Vb, nullptr, 32768, 512, 512, 512, 512, 512,
      0, 0, 0, 0, 0, 0);
  // 4) L2 norms of q,k channels (over HW)
  col_l2norm<<<4096, 256, 0, stream>>>(Qb, nq);
  col_l2norm<<<4096, 256, 0, stream>>>(Kb, nk);
  // 5) channel attention + softmax per (b,h)
  attn_softmax<<<64, 128, 0, stream>>>(Qb, Kb, nq, nk, P);
  // 6) Y_head = V_head * P^T, batched over z = b*8+h (K=64 -> one tile)
  gemm_xwT<0><<<dim3(32, 1, 64), 256, 0, stream>>>(
      Vb, P, Y, nullptr, 4096, 64, 64, 512, 64, 512,
      (long)4096 * 512, 64,            // A offsets: b, h
      (long)8 * 4096, 4096,            // W (P) offsets: z*4096 decomposed
      (long)4096 * 512, 64);           // C offsets: b, h
  // 7) projection + NCHW fp32 scatter into d_out
  gemm_xwT<1><<<dim3(256, 8, 1), 256, 0, stream>>>(
      Y, Wb + 3 * 262144, nullptr, out, 32768, 512, 512, 512, 512, 512,
      0, 0, 0, 0, 0, 0);
}
